// _NonLocalNd_bn_25967372271684
// MI455X (gfx1250) — matrix operand order repaired and checked by reading, not compiled, not run on hardware
//
#include <hip/hip_runtime.h>
#include <hip/hip_bf16.h>
#include <math.h>

// ---------------------------------------------------------------------------
// NonLocal block for MI455X (gfx1250, wave32, WMMA bf16 16x16x32, f32 accum)
// Async global->LDS staging (ASYNCcnt) + double-buffered LDS tiles.
// ---------------------------------------------------------------------------

typedef __attribute__((ext_vector_type(16))) __bf16 v16bf;
typedef __attribute__((ext_vector_type(8)))  __bf16 v8bf;
typedef __attribute__((ext_vector_type(4)))  __bf16 v4bf;
typedef __attribute__((ext_vector_type(8)))  float  v8f;

// Problem constants
#define Bn 16
#define Cn 1024
#define Pn 512
#define Nn 4096   // 64*64
#define Mn 1024   // 32*32
#define INV_T 0.883883476483184f   // 1/(sqrt(512)*0.05)

// ---- workspace layout (bytes) ---------------------------------------------
#define OFF_XC    ((size_t)0)                      // bf16 [B][C][N] centered x   (134MB)  -> reused by simg
#define OFF_IXC   ((size_t)134217728)              // bf16 [B][C][M] centered ix  (33.5MB)
#define OFF_IXR   ((size_t)167772160)              // bf16 [B][C][M] raw ix       (33.5MB)
#define OFF_QT    ((size_t)201326592)              // bf16 [B][N][P] q^T          (67MB)   -> reused by outsim^T
#define OFF_KT    ((size_t)268435456)              // bf16 [B][M][P] k^T          (16.8MB)
#define OFF_V     ((size_t)285212672)              // bf16 [B][P][M] v            (16.8MB)
#define OFF_VGC   ((size_t)301989888)              // bf16 [B][P][M] vgc          (16.8MB)
#define OFF_SIM   ((size_t)318767104)              // f32  [B][N][M] raw logits   (268MB)
#define OFF_WQB   ((size_t)587202560)              // bf16 weights
#define OFF_WKB   ((size_t)588251136)
#define OFF_WVB   ((size_t)589299712)
#define OFF_WGCB  ((size_t)590348288)
#define OFF_WOUTB ((size_t)591396864)
#define OFF_MASK  ((size_t)592445440)              // f32 [B][M]
#define OFF_GCB   ((size_t)592510976)              // f32 [B][P]  beta * vgc@mask
#define OFF_GCP   ((size_t)592543744)              // f32 [B][C]  wout @ gcb
#define OFF_SIMG  OFF_XC                           // bf16 [B][N][M] gamma*softmax
#define OFF_OUTS  OFF_QT                           // bf16 [B][N][P] (sim@v^T)^T

// ---------------------------------------------------------------------------
// Async 16B copy global -> LDS (CDNA5 GLOBAL_LOAD_ASYNC_TO_LDS_B128).
// LDS offset = low 32 bits of the generic pointer (ISA 10.2: LDS aperture
// uses the low 32 address bits as the LDS address).
// ---------------------------------------------------------------------------
__device__ __forceinline__ void async_cp16(void* lds, const void* g)
{
  unsigned loff = (unsigned)(size_t)lds;
  asm volatile("global_load_async_to_lds_b128 %0, %1, off"
               :: "v"(loff), "v"(g) : "memory");
}
__device__ __forceinline__ void wait_async0()
{
  asm volatile("s_wait_asynccnt 0x0" ::: "memory");
}

// ---------------------------------------------------------------------------
// Fragment loads from LDS tiles stored as [row][40] bf16 (pad 8 -> 80B stride)
// A frag (16x32, 16-bit): lane<16: row=lane, K 0-7 & 16-23 ; lane>=16: K 8-15 & 24-31
// B frag (32x16, 16-bit): lane<16: col=lane, K 0-7 & 16-23 ; lane>=16: K 8-15 & 24-31
// ---------------------------------------------------------------------------
__device__ __forceinline__ v16bf frag_A(const __bf16* tile, int lane) {
  int r  = lane & 15;
  int kb = (lane >> 4) << 3;                 // 0 or 8
  v8bf lo = *(const v8bf*)(tile + r * 40 + kb);
  v8bf hi = *(const v8bf*)(tile + r * 40 + kb + 16);
  return __builtin_shufflevector(lo, hi, 0,1,2,3,4,5,6,7,8,9,10,11,12,13,14,15);
}
__device__ __forceinline__ v16bf frag_B(const __bf16* tile, int lane) {
  int c  = lane & 15;
  int k0 = (lane >> 4) << 3;                 // 0 or 8
  v8bf lo = *(const v8bf*)(tile + c * 40 + k0);
  v8bf hi = *(const v8bf*)(tile + c * 40 + k0 + 16);
  return __builtin_shufflevector(lo, hi, 0,1,2,3,4,5,6,7,8,9,10,11,12,13,14,15);
}

// ---------------------------------------------------------------------------
// Generic bf16 GEMM:  Y[r,c] = sum_k A[r,k] * B'[k,c]
//   A: row-major [R][K] (batched via sA)
//   B: TRB ? stored [K][Cc] (transpose-staged) : stored [Cc][K] (K-contiguous)
//   EPI 0: bf16 out                     (STT: store transposed [Cc][R])
//   EPI 1: bf16 out + bias[row]
//   EPI 2: f32 out
//   EPI 3: f32 out + gcp[b*R+row] + resid[b*sY + row*Cc + col]   (final)
// Tile: workgroup 128x64 (8 waves of 32x32), K step 32, double-buffered LDS.
// ---------------------------------------------------------------------------
template<bool TRB, bool STT, int EPI>
__global__ __launch_bounds__(256)
void gemm_bf16(const __bf16* __restrict__ A, const __bf16* __restrict__ Bm,
               void* __restrict__ Yv, int R, int Cc, int K,
               size_t sA, size_t sB, size_t sY,
               const float* __restrict__ bias,
               const float* __restrict__ gcp,
               const float* __restrict__ resid)
{
  __shared__ __align__(16) __bf16 At[2][128 * 40];
  __shared__ __align__(16) __bf16 Bt[2][64 * 40];

  const int t    = threadIdx.x;
  const int lane = t & 31;
  const int wv   = t >> 5;
  const int wr   = wv & 3;          // wave row 0..3  (32 rows each)
  const int wc   = wv >> 2;         // wave col 0..1  (32 cols each)
  const int b    = blockIdx.z;
  const int c0   = blockIdx.x * 64;

  const __bf16* Ag = A + (size_t)b * sA + (size_t)blockIdx.y * 128 * K;
  const __bf16* Bg = Bm + (size_t)b * sB;

  // per-thread staging coordinates
  const int arow0 = t >> 2;                 // A chunk 0: rows 0..63
  const int ac8   = (t & 3) << 3;
  const int brow  = t >> 2;                 // B direct: rows 0..63
  const int tcc4  = (t & 15) << 2;          // B transpose: 4 cols
  const int tkk   = t >> 4;                 // B transpose: k within 0..15

  auto stage = [&](int buf, int k0) {
    // A (128x32): 2 async 16B chunks per thread
    async_cp16(&At[buf][arow0 * 40 + ac8],
               Ag + (size_t)arow0 * K + k0 + ac8);
    async_cp16(&At[buf][(arow0 + 64) * 40 + ac8],
               Ag + (size_t)(arow0 + 64) * K + k0 + ac8);
    if (!TRB) {
      async_cp16(&Bt[buf][brow * 40 + ac8],
                 Bg + (size_t)(c0 + brow) * K + k0 + ac8);
    } else {
      // B stored [K][Cc]: 8B loads along contiguous cols, transpose into LDS
#pragma unroll
      for (int j = 0; j < 2; ++j) {
        int kk = tkk + j * 16;
        const __bf16* src = Bg + (size_t)(k0 + kk) * Cc + c0 + tcc4;
        v4bf d = *(const v4bf*)src;
        if (k0 + 32 < K) __builtin_prefetch(src + 32 * Cc, 0, 3);
#pragma unroll
        for (int i = 0; i < 4; ++i) Bt[buf][(tcc4 + i) * 40 + kk] = d[i];
      }
    }
  };

  v8f acc[2][2] = {};

  stage(0, 0);
  for (int k0 = 0; k0 < K; k0 += 32) {
    const int buf = (k0 >> 5) & 1;
    wait_async0();                 // own async copies into buf complete
    __syncthreads();               // all waves staged buf / done reading buf^1
    if (k0 + 32 < K) stage(buf ^ 1, k0 + 32);

    const __bf16* Ab = &At[buf][0];
    const __bf16* Bb = &Bt[buf][0];
    v16bf af0 = frag_A(Ab + (wr * 32     ) * 40, lane);
    v16bf af1 = frag_A(Ab + (wr * 32 + 16) * 40, lane);
    v16bf bf0 = frag_B(Bb + (wc * 32     ) * 40, lane);
    v16bf bf1 = frag_B(Bb + (wc * 32 + 16) * 40, lane);

    acc[0][0] = __builtin_amdgcn_wmma_f32_16x16x32_bf16(false, af0, false, bf0, (short)0, acc[0][0], false, false);
    acc[0][1] = __builtin_amdgcn_wmma_f32_16x16x32_bf16(false, af0, false, bf1, (short)0, acc[0][1], false, false);
    acc[1][0] = __builtin_amdgcn_wmma_f32_16x16x32_bf16(false, af1, false, bf0, (short)0, acc[1][0], false, false);
    acc[1][1] = __builtin_amdgcn_wmma_f32_16x16x32_bf16(false, af1, false, bf1, (short)0, acc[1][1], false, false);
  }

  // ---- epilogue ----
  const int rbase = blockIdx.y * 128 + wr * 32;
#pragma unroll
  for (int i = 0; i < 2; ++i) {
#pragma unroll
    for (int j = 0; j < 2; ++j) {
      int rb  = rbase + i * 16 + ((lane >> 4) << 3);   // first of 8 consecutive rows
      int col = c0 + wc * 32 + j * 16 + (lane & 15);
      if (STT) {
        __bf16* Yt = (__bf16*)Yv + (size_t)b * sY + (size_t)col * R + rb;
        v8bf pk;
#pragma unroll
        for (int e = 0; e < 8; ++e) pk[e] = (__bf16)acc[i][j][e];
        *(v8bf*)Yt = pk;
      } else if (EPI == 0 || EPI == 1) {
        __bf16* Yp = (__bf16*)Yv + (size_t)b * sY;
#pragma unroll
        for (int e = 0; e < 8; ++e) {
          int row = rb + e;
          float v = acc[i][j][e];
          if (EPI == 1) v += bias[row];
          Yp[(size_t)row * Cc + col] = (__bf16)v;
        }
      } else if (EPI == 2) {
        float* Yp = (float*)Yv + (size_t)b * sY;
#pragma unroll
        for (int e = 0; e < 8; ++e)
          Yp[(size_t)(rb + e) * Cc + col] = acc[i][j][e];
      } else {   // EPI == 3 : final  (+gc proj, +residual)
        float* Yp = (float*)Yv + (size_t)b * sY;
#pragma unroll
        for (int e = 0; e < 8; ++e) {
          int row = rb + e;
          size_t ix = (size_t)row * Cc + col;
          Yp[ix] = acc[i][j][e] + gcp[b * R + row] + resid[(size_t)b * sY + ix];
        }
      }
    }
  }
}

// ---------------------------------------------------------------------------
// Pool + center + bf16 convert.  One block per (b,c): N=4096 elems, M=1024.
// q - mean(q) == wq @ (x - mean(x))  -> center inputs, biases cancel.
// ---------------------------------------------------------------------------
__global__ __launch_bounds__(256)
void pool_center(const float* __restrict__ x, __bf16* __restrict__ xc,
                 __bf16* __restrict__ ixc, __bf16* __restrict__ ixr)
{
  const int bc = blockIdx.x;
  const int t  = threadIdx.x;
  const float* xp = x + (size_t)bc * Nn;

  float v[16];
  float s = 0.f;
#pragma unroll
  for (int i = 0; i < 16; ++i) { v[i] = xp[t * 16 + i]; s += v[i]; }

  float mx[4];
  float sp = 0.f;
#pragma unroll
  for (int j = 0; j < 4; ++j) {
    int qi = t + j * 256;
    int oy = qi >> 5, ox = qi & 31;
    const float* r0 = xp + (oy * 2) * 64 + ox * 2;
    float m = fmaxf(fmaxf(r0[0], r0[1]), fmaxf(r0[64], r0[65]));
    mx[j] = m; sp += m;
  }

  __shared__ float red[256], red2[256];
  red[t] = s; red2[t] = sp; __syncthreads();
  for (int off = 128; off > 0; off >>= 1) {
    if (t < off) { red[t] += red[t + off]; red2[t] += red2[t + off]; }
    __syncthreads();
  }
  const float xmean  = red[0]  * (1.f / 4096.f);
  const float ixmean = red2[0] * (1.f / 1024.f);

  __bf16* xcp = xc + (size_t)bc * Nn;
#pragma unroll
  for (int i = 0; i < 16; ++i) xcp[t * 16 + i] = (__bf16)(v[i] - xmean);

  __bf16* a1 = ixc + (size_t)bc * Mn;
  __bf16* a2 = ixr + (size_t)bc * Mn;
#pragma unroll
  for (int j = 0; j < 4; ++j) {
    int qi = t + j * 256;
    a1[qi] = (__bf16)(mx[j] - ixmean);
    a2[qi] = (__bf16)mx[j];
  }
}

__global__ __launch_bounds__(256)
void cvt_f32_bf16(const float* __restrict__ s, __bf16* __restrict__ d, int n)
{
  int i = blockIdx.x * 256 + threadIdx.x;
  if (i < n) d[i] = (__bf16)s[i];
}

// mask[b,m] = sum_c wmask[c]*ix[b,c,m] + bmask
__global__ __launch_bounds__(256)
void mask_linear(const __bf16* __restrict__ ixr, const float* __restrict__ wmask,
                 const float* __restrict__ bmask, float* __restrict__ mask)
{
  __shared__ float wsm[Cn];
  int t = threadIdx.x;
  for (int c = t; c < Cn; c += 256) wsm[c] = wmask[c];
  __syncthreads();
  int idx = blockIdx.x * 256 + t;
  int b = idx >> 10, m = idx & 1023;
  const __bf16* p = ixr + ((size_t)b * Cn) * Mn + m;
  float s = bmask[0];
#pragma unroll 8
  for (int c = 0; c < Cn; ++c) s += wsm[c] * (float)p[(size_t)c * Mn];
  mask[idx] = s;
}

// softmax over M per batch, in place
__global__ __launch_bounds__(256)
void mask_softmax(float* __restrict__ mask)
{
  __shared__ float red[256];
  int t = threadIdx.x;
  float* p = mask + (size_t)blockIdx.x * Mn;
  float v[4], mx = -1e30f;
#pragma unroll
  for (int i = 0; i < 4; ++i) { v[i] = p[t * 4 + i]; mx = fmaxf(mx, v[i]); }
  red[t] = mx; __syncthreads();
  for (int off = 128; off > 0; off >>= 1) { if (t < off) red[t] = fmaxf(red[t], red[t + off]); __syncthreads(); }
  mx = red[0]; __syncthreads();
  float s = 0.f;
#pragma unroll
  for (int i = 0; i < 4; ++i) { v[i] = __expf(v[i] - mx); s += v[i]; }
  red[t] = s; __syncthreads();
  for (int off = 128; off > 0; off >>= 1) { if (t < off) red[t] += red[t + off]; __syncthreads(); }
  float inv = 1.f / red[0];
#pragma unroll
  for (int i = 0; i < 4; ++i) p[t * 4 + i] = v[i] * inv;
}

// fused: softmax over M of raw logits (with temp scale inside exp), * gamma -> bf16
__global__ __launch_bounds__(256)
void sim_softmax(const float* __restrict__ sim, __bf16* __restrict__ simg,
                 const float* __restrict__ gamma)
{
  __shared__ float red[256];
  int t = threadIdx.x;
  const float* rp = sim + (size_t)blockIdx.x * Mn;
  float4 v = ((const float4*)rp)[t];
  float mx = fmaxf(fmaxf(v.x, v.y), fmaxf(v.z, v.w));
  red[t] = mx; __syncthreads();
  for (int off = 128; off > 0; off >>= 1) { if (t < off) red[t] = fmaxf(red[t], red[t + off]); __syncthreads(); }
  mx = red[0]; __syncthreads();
  float e0 = __expf((v.x - mx) * INV_T);
  float e1 = __expf((v.y - mx) * INV_T);
  float e2 = __expf((v.z - mx) * INV_T);
  float e3 = __expf((v.w - mx) * INV_T);
  red[t] = e0 + e1 + e2 + e3; __syncthreads();
  for (int off = 128; off > 0; off >>= 1) { if (t < off) red[t] += red[t + off]; __syncthreads(); }
  float w = gamma[0] / red[0];
  __bf16* op = simg + (size_t)blockIdx.x * Mn + t * 4;
  op[0] = (__bf16)(e0 * w); op[1] = (__bf16)(e1 * w);
  op[2] = (__bf16)(e2 * w); op[3] = (__bf16)(e3 * w);
}

// gcb[b,p] = beta * sum_m vgc[b,p,m] * mask[b,m]
__global__ __launch_bounds__(256)
void gc_reduce(const __bf16* __restrict__ vgc, const float* __restrict__ mask,
               const float* __restrict__ beta, float* __restrict__ gcb)
{
  __shared__ float red[256];
  int t = threadIdx.x;
  int row = blockIdx.x;              // b*P + p
  int b = row >> 9;
  const __bf16* vp = vgc + (size_t)row * Mn;
  const float*  mp = mask + (size_t)b * Mn;
  float s = 0.f;
#pragma unroll
  for (int i = 0; i < 4; ++i) { int m = t + i * 256; s += (float)vp[m] * mp[m]; }
  red[t] = s; __syncthreads();
  for (int off = 128; off > 0; off >>= 1) { if (t < off) red[t] += red[t + off]; __syncthreads(); }
  if (t == 0) gcb[row] = beta[0] * red[0];
}

// gcproj[b,c] = sum_p wout[c,p] * gcb[b,p]     (f32 weights for accuracy)
__global__ __launch_bounds__(256)
void gc_project(const float* __restrict__ wout, const float* __restrict__ gcb,
                float* __restrict__ gcp)
{
  __shared__ float red[256];
  int t = threadIdx.x;
  int row = blockIdx.x;              // b*C + c
  int b = row >> 10, c = row & 1023;
  const float* wp = wout + (size_t)c * Pn;
  const float* gp = gcb + (size_t)b * Pn;
  float s = wp[t] * gp[t] + wp[t + 256] * gp[t + 256];
  red[t] = s; __syncthreads();
  for (int off = 128; off > 0; off >>= 1) { if (t < off) red[t] += red[t + off]; __syncthreads(); }
  if (t == 0) gcp[row] = red[0];
}

// ---------------------------------------------------------------------------
extern "C" void kernel_launch(void* const* d_in, const int* in_sizes, int n_in,
                              void* d_out, int out_size, void* d_ws, size_t ws_size,
                              hipStream_t stream)
{
  (void)in_sizes; (void)n_in; (void)out_size; (void)ws_size;
  const float* x     = (const float*)d_in[0];
  const float* wq    = (const float*)d_in[1];
  const float* wk    = (const float*)d_in[3];
  const float* wv    = (const float*)d_in[5];
  const float* bv    = (const float*)d_in[6];
  const float* wgc   = (const float*)d_in[7];
  const float* bgc   = (const float*)d_in[8];
  const float* wmask = (const float*)d_in[9];
  const float* bmask = (const float*)d_in[10];
  const float* wout  = (const float*)d_in[11];
  const float* gamma = (const float*)d_in[12];
  const float* beta  = (const float*)d_in[13];
  float* out = (float*)d_out;

  char* ws = (char*)d_ws;
  __bf16* xc    = (__bf16*)(ws + OFF_XC);
  __bf16* ixc   = (__bf16*)(ws + OFF_IXC);
  __bf16* ixr   = (__bf16*)(ws + OFF_IXR);
  __bf16* qt    = (__bf16*)(ws + OFF_QT);
  __bf16* kt    = (__bf16*)(ws + OFF_KT);
  __bf16* v     = (__bf16*)(ws + OFF_V);
  __bf16* vgc   = (__bf16*)(ws + OFF_VGC);
  float*  sim   = (float*) (ws + OFF_SIM);
  __bf16* simg  = (__bf16*)(ws + OFF_SIMG);
  __bf16* outs  = (__bf16*)(ws + OFF_OUTS);
  __bf16* wqb   = (__bf16*)(ws + OFF_WQB);
  __bf16* wkb   = (__bf16*)(ws + OFF_WKB);
  __bf16* wvb   = (__bf16*)(ws + OFF_WVB);
  __bf16* wgcb  = (__bf16*)(ws + OFF_WGCB);
  __bf16* woutb = (__bf16*)(ws + OFF_WOUTB);
  float*  maskb = (float*) (ws + OFF_MASK);
  float*  gcb   = (float*) (ws + OFF_GCB);
  float*  gcp   = (float*) (ws + OFF_GCP);

  const int WN = Pn * Cn;   // 524288
  cvt_f32_bf16<<<WN / 256, 256, 0, stream>>>(wq,   wqb,   WN);
  cvt_f32_bf16<<<WN / 256, 256, 0, stream>>>(wk,   wkb,   WN);
  cvt_f32_bf16<<<WN / 256, 256, 0, stream>>>(wv,   wvb,   WN);
  cvt_f32_bf16<<<WN / 256, 256, 0, stream>>>(wgc,  wgcb,  WN);
  cvt_f32_bf16<<<WN / 256, 256, 0, stream>>>(wout, woutb, WN);

  pool_center<<<Bn * Cn, 256, 0, stream>>>(x, xc, ixc, ixr);

  // q^T = (wq @ xc)^T            [B][N][P]
  gemm_bf16<true, true, 0><<<dim3(Nn / 64, Pn / 128, Bn), 256, 0, stream>>>(
      wqb, xc, qt, Pn, Nn, Cn, 0, (size_t)Cn * Nn, (size_t)Nn * Pn, nullptr, nullptr, nullptr);
  // k^T = (wk @ ixc)^T           [B][M][P]
  gemm_bf16<true, true, 0><<<dim3(Mn / 64, Pn / 128, Bn), 256, 0, stream>>>(
      wkb, ixc, kt, Pn, Mn, Cn, 0, (size_t)Cn * Mn, (size_t)Mn * Pn, nullptr, nullptr, nullptr);
  // v = wv @ ix + bv             [B][P][M]
  gemm_bf16<true, false, 1><<<dim3(Mn / 64, Pn / 128, Bn), 256, 0, stream>>>(
      wvb, ixr, v, Pn, Mn, Cn, 0, (size_t)Cn * Mn, (size_t)Pn * Mn, bv, nullptr, nullptr);
  // vgc = wgc @ ix + bgc         [B][P][M]
  gemm_bf16<true, false, 1><<<dim3(Mn / 64, Pn / 128, Bn), 256, 0, stream>>>(
      wgcb, ixr, vgc, Pn, Mn, Cn, 0, (size_t)Cn * Mn, (size_t)Pn * Mn, bgc, nullptr, nullptr);

  // sim = q^T k                  [B][N][M]  f32
  gemm_bf16<false, false, 2><<<dim3(Mn / 64, Nn / 128, Bn), 256, 0, stream>>>(
      qt, kt, sim, Nn, Mn, Pn, (size_t)Nn * Pn, (size_t)Mn * Pn, (size_t)Nn * Mn,
      nullptr, nullptr, nullptr);

  // gamma * softmax(sim / (scale*temp))  -> bf16 (reuses xc buffer)
  sim_softmax<<<Bn * Nn, 256, 0, stream>>>(sim, simg, gamma);

  // outsim^T = (v @ simg^T)^T    [B][N][P] (reuses qt buffer)
  gemm_bf16<false, true, 0><<<dim3(Nn / 64, Pn / 128, Bn), 256, 0, stream>>>(
      v, simg, outs, Pn, Nn, Mn, (size_t)Pn * Mn, (size_t)Nn * Mn, (size_t)Nn * Pn,
      nullptr, nullptr, nullptr);

  // global-context branch (small)
  mask_linear<<<Bn * Mn / 256, 256, 0, stream>>>(ixr, wmask, bmask, maskb);
  mask_softmax<<<Bn, 256, 0, stream>>>(maskb);
  gc_reduce<<<Bn * Pn, 256, 0, stream>>>(vgc, maskb, beta, gcb);
  gc_project<<<Bn * Cn, 256, 0, stream>>>(wout, gcb, gcp);

  // out = wout @ outsim + gcproj + residual   [B][C][N] f32 -> d_out
  gemm_bf16<false, false, 3><<<dim3(Nn / 64, Cn / 128, Bn), 256, 0, stream>>>(
      woutb, outs, out, Cn, Nn, Pn, 0, (size_t)Nn * Pn, (size_t)Cn * Nn,
      nullptr, gcp, x);
}
